// MultiScaleAttention_32418413150327
// MI455X (gfx1250) — matrix operand order repaired and checked by reading, not compiled, not run on hardware
//
#include <hip/hip_runtime.h>
#include <hip/hip_bf16.h>
#include <cstdint>

// ---------------------------------------------------------------------------
// MultiScaleAttention on MI455X (gfx1250): all GEMM-shaped work routed through
// v_wmma_f32_16x16x32_bf16 (bf16 inputs, f32 accumulate). The two large
// projections stage their shared A tile into LDS with
// global_load_async_to_lds_b128 (ASYNCcnt) and read it back with ds_load.
// ---------------------------------------------------------------------------

typedef __attribute__((ext_vector_type(16))) __bf16 v16bf;
typedef __attribute__((ext_vector_type(8)))  __bf16 v8bf;
typedef __attribute__((ext_vector_type(8)))  float  v8f;

typedef __attribute__((address_space(3))) const v8bf* lds_v8bf_cp;

#define BATCH   8
#define NTOK    4096      // 64*64 tokens per image
#define CDIM    512
#define NHEADS  8
#define HD      64
#define IMGW    64
#define MTOT    (BATCH*NTOK)   // 32768 rows
#define ATTN_SCALE 0.125f      // HD^-0.5
#define A_STRIDE 520           // padded LDS row stride (bf16 elems) -> 4-bank skew

// ---------------------------------------------------------------------------
// WMMA + fragment helpers
// ---------------------------------------------------------------------------
__device__ __forceinline__ v8f wmma_bf16(v16bf a, v16bf b, v8f c) {
  return __builtin_amdgcn_wmma_f32_16x16x32_bf16(false, a, false, b,
                                                 (short)0, c, false, false);
}

// A-matrix 16x32 bf16 fragment: lane half hb = lane>>4 selects K sub-block.
__device__ __forceinline__ int frag_k_a(int lane, int e) {
  int hb = (lane >> 4) & 1;
  return (e < 8) ? (hb * 8 + e) : (16 + hb * 8 + (e - 8));
}
// B-matrix 32x16 bf16 fragment: lane n' = lane&15 is the column,
// element e -> k = hb*8 + e (e<8), 16 + hb*8 + (e-8) (e>=8), same as A.
__device__ __forceinline__ int frag_k_b(int lane, int e) {
  int hb = (lane >> 4) & 1;
  return (e < 8) ? (hb * 8 + e) : (16 + hb * 8 + (e - 8));
}

// Load A fragment from a row-major row in global memory (K contiguous).
__device__ __forceinline__ v16bf frag_load_a(const __bf16* rowptr) {
  int hb = (threadIdx.x >> 4) & 1;
  v8bf lo = *(const v8bf*)(rowptr + hb * 8);
  v8bf hi = *(const v8bf*)(rowptr + 16 + hb * 8);
  v16bf f;
#pragma unroll
  for (int e = 0; e < 8; ++e) { f[e] = lo[e]; f[8 + e] = hi[e]; }
  return f;
}
// Same, but from LDS (byte address, wave-relative) -> ds_load_b128 x2.
__device__ __forceinline__ v16bf frag_load_a_lds(unsigned byteaddr) {
  int hb = (threadIdx.x >> 4) & 1;
  v8bf lo = *(lds_v8bf_cp)(uintptr_t)(byteaddr + (unsigned)(hb * 16));
  v8bf hi = *(lds_v8bf_cp)(uintptr_t)(byteaddr + 32u + (unsigned)(hb * 16));
  v16bf f;
#pragma unroll
  for (int e = 0; e < 8; ++e) { f[e] = lo[e]; f[8 + e] = hi[e]; }
  return f;
}
// Load B fragment for this lane's column from a K-contiguous row of B^T.
__device__ __forceinline__ v16bf frag_load_b(const __bf16* colptr) {
  int hb = (threadIdx.x >> 4) & 1;
  v8bf lo = *(const v8bf*)(colptr + hb * 8);
  v8bf hi = *(const v8bf*)(colptr + 16 + hb * 8);
  v16bf f;
#pragma unroll
  for (int e = 0; e < 8; ++e) { f[e] = lo[e]; f[8 + e] = hi[e]; }
  return f;
}

// Async global->LDS copy (16B per lane), tracked by ASYNCcnt.
__device__ __forceinline__ void async_load_b128(unsigned lds_off, const void* gaddr) {
  asm volatile("global_load_async_to_lds_b128 %0, %1, off"
               :: "v"(lds_off), "v"(gaddr) : "memory");
}
__device__ __forceinline__ void async_wait0() {
  asm volatile("s_wait_asynccnt 0x0" ::: "memory");
}

// Token index inside the 64x64 image for (patch p, in-patch index idx).
__device__ __forceinline__ int tok_of(int p, int idx, int ps) {
  int wpat = IMGW / ps;
  int hp = p / wpat, wp = p % wpat;
  int dy = idx / ps, dx = idx % ps;
  return (hp * ps + dy) * IMGW + wp * ps + dx;
}

// ---------------------------------------------------------------------------
// Cast / pack kernels
// ---------------------------------------------------------------------------
__global__ void k_cast_bf16(const float* __restrict__ in, __bf16* __restrict__ out, int n) {
  int i = blockIdx.x * blockDim.x + threadIdx.x;
  if (i < n) out[i] = (__bf16)in[i];
}

// out[c][r] = in[r][c]  (cast f32 -> bf16)
__global__ void k_transpose_cast(const float* __restrict__ in, __bf16* __restrict__ out,
                                 int rows, int cols) {
  int i = blockIdx.x * blockDim.x + threadIdx.x;
  if (i < rows * cols) {
    int r = i / cols, c = i % cols;
    out[(size_t)c * rows + r] = (__bf16)in[i];
  }
}

// Fused [Wq | Wkv] transposed: out[n][k], n in [0,1536)
__global__ void k_pack_wqkvT(const float* __restrict__ Wq, const float* __restrict__ Wkv,
                             __bf16* __restrict__ out) {
  int i = blockIdx.x * blockDim.x + threadIdx.x;
  if (i < 1536 * 512) {
    int n = i / 512, k = i % 512;
    float v = (n < 512) ? Wq[(size_t)k * 512 + n] : Wkv[(size_t)k * 1024 + (n - 512)];
    out[i] = (__bf16)v;
  }
}

// ---------------------------------------------------------------------------
// QKV projection: y = x @ [Wq|Wkv], scattered to q/k/v [B][H][N][HD] bf16.
// Block (4 waves) shares one 16-row A tile staged async into LDS; each wave
// computes a 16(M) x 64(N) tile over its own weight columns.
// ---------------------------------------------------------------------------
__global__ void k_gemm_qkv(const __bf16* __restrict__ xb, const __bf16* __restrict__ wT,
                           __bf16* __restrict__ qb, __bf16* __restrict__ kb,
                           __bf16* __restrict__ vb) {
  __shared__ __bf16 smemA[16 * A_STRIDE];
  const int NGRP = (3 * CDIM) / 256;           // 6 groups of 256 columns
  int mt = blockIdx.x / NGRP;
  int ng = blockIdx.x % NGRP;
  int w = threadIdx.x >> 5;
  int lane = threadIdx.x & 31;

  // Stage A tile (16 x 512 bf16) into LDS with async copies.
  const __bf16* aBase = xb + (size_t)mt * 16 * CDIM;
  unsigned ldsBase = (unsigned)(uintptr_t)&smemA[0];
#pragma unroll
  for (int i = 0; i < 8; ++i) {
    int c = (int)threadIdx.x + i * 128;        // 1024 chunks of 8 elems
    int row = c >> 6, col = (c & 63) << 3;
    async_load_b128(ldsBase + (unsigned)(row * A_STRIDE + col) * 2u,
                    aBase + (size_t)row * CDIM + col);
  }
  async_wait0();
  __syncthreads();

  int nt = ng * 4 + w;
  int ncol = nt * 64 + (lane & 15);
  unsigned arow = ldsBase + (unsigned)((lane & 15) * A_STRIDE) * 2u;
  v8f acc[4] = {};
  for (int k0 = 0; k0 < CDIM; k0 += 32) {
    v16bf af = frag_load_a_lds(arow + (unsigned)(k0 * 2));
#pragma unroll
    for (int j = 0; j < 4; ++j) {
      v16bf bf = frag_load_b(wT + (size_t)(ncol + j * 16) * CDIM + k0);
      acc[j] = wmma_bf16(af, bf, acc[j]);
    }
  }
#pragma unroll
  for (int j = 0; j < 4; ++j) {
    int nc = ncol + j * 16;
    __bf16* dst; int head, hd;
    if (nc < CDIM) { dst = qb; head = nc >> 6; hd = nc & 63; }
    else { int t = nc - CDIM; dst = (t >> 9) ? vb : kb; head = (t >> 6) & 7; hd = t & 63; }
#pragma unroll
    for (int r = 0; r < 8; ++r) {
      int m = mt * 16 + ((lane >> 4) << 3) + r;
      int b = m >> 12, tok = m & (NTOK - 1);
      dst[(((size_t)b * NHEADS + head) * NTOK + tok) * HD + hd] = (__bf16)acc[j][r];
    }
  }
}

// ---------------------------------------------------------------------------
// corr = (qp @ kp^T) * scale  [+ prev corr for same-ps heads]
// n = ps*ps in {4,16,64,256}. For n==4: 4 patches block-diagonal per tile.
// ---------------------------------------------------------------------------
__global__ void k_corr(const __bf16* __restrict__ qb, const __bf16* __restrict__ kb,
                       float* __restrict__ corrOut, const float* __restrict__ corrPrev,
                       int head, int ps, int addPrev) {
  int n = ps * ps;
  int P = NTOK / n;
  int T = (n >= 16) ? n / 16 : 1;
  int Q = (n >= 16) ? 1 : 16 / n;
  int pgc = P / Q;
  int tpb = pgc * T * T;
  int gw = blockIdx.x * (blockDim.x >> 5) + (threadIdx.x >> 5);
  if (gw >= BATCH * tpb) return;
  int b = gw / tpb, rem = gw % tpb;
  int pg = rem / (T * T), tt = rem % (T * T);
  int ti = tt / T, tj = tt % T;
  int lane = threadIdx.x & 31;
  int am = lane & 15;

  int a_patch, a_local, b_patch, b_local;
  if (Q == 1) { a_patch = pg; a_local = ti * 16 + am; b_patch = pg; b_local = tj * 16 + am; }
  else        { a_patch = pg * Q + am / n; a_local = am % n; b_patch = a_patch; b_local = a_local; }

  int atok = tok_of(a_patch, a_local, ps);
  int btok = tok_of(b_patch, b_local, ps);
  const __bf16* abase = qb + (((size_t)b * NHEADS + head) * NTOK + atok) * HD;
  const __bf16* bbase = kb + (((size_t)b * NHEADS + head) * NTOK + btok) * HD;

  v8f acc = {};
#pragma unroll
  for (int k0 = 0; k0 < HD; k0 += 32) {
    v16bf af = frag_load_a(abase + k0);
    v16bf bf = frag_load_b(bbase + k0);
    acc = wmma_bf16(af, bf, acc);
  }

  float* cb = corrOut + (size_t)b * NTOK * n;
  const float* pb = addPrev ? (corrPrev + (size_t)b * NTOK * n) : nullptr;
#pragma unroll
  for (int r = 0; r < 8; ++r) {
    int m = ((lane >> 4) << 3) + r;
    int rp, rl, jl;
    if (Q == 1) { rp = pg; rl = ti * 16 + m; jl = tj * 16 + am; }
    else {
      rp = pg * Q + m / n; rl = m % n; jl = b_local;
      if (rp != b_patch) continue;   // off-diagonal cross-patch block: discard
    }
    size_t o = ((size_t)rp * n + rl) * n + jl;
    float v = acc[r] * ATTN_SCALE;
    if (addPrev) v += pb[o];
    cb[o] = v;
  }
}

// ---------------------------------------------------------------------------
// Cross-scale mixing (1x1 conv) as per-batch GEMM:
//   out[b][o][s] = sum_k Wc[o][k] * cat[b][k][s] + bc[o]
// ---------------------------------------------------------------------------
__global__ void k_mix(const __bf16* __restrict__ wcb, const float* __restrict__ bc,
                      const float* __restrict__ raw, const float* __restrict__ prevSmall,
                      float* __restrict__ outc, int Pn, int E, int S) {
  int Kc = Pn + E;
  int mtc = Pn / 16, stc = S / 16;
  int gw = blockIdx.x * (blockDim.x >> 5) + (threadIdx.x >> 5);
  if (gw >= BATCH * mtc * stc) return;
  int b = gw / (mtc * stc);
  int r2 = gw % (mtc * stc);
  int mt = r2 / stc, st = r2 % stc;
  int lane = threadIdx.x & 31;
  int am = mt * 16 + (lane & 15);
  int sn = st * 16 + (lane & 15);
  const float* catRaw = raw + (size_t)b * Pn * S;
  const float* catExt = prevSmall + (size_t)b * E * S;

  v8f acc = {};
  for (int k0 = 0; k0 < Kc; k0 += 32) {
    v16bf af = {}, bf = {};
#pragma unroll
    for (int e = 0; e < 16; ++e) {
      int ka = k0 + frag_k_a(lane, e);
      af[e] = (ka < Kc) ? wcb[(size_t)am * Kc + ka] : (__bf16)0.f;
      int kbb = k0 + frag_k_b(lane, e);
      float cv = 0.f;
      if (kbb < Pn)      cv = catRaw[(size_t)kbb * S + sn];
      else if (kbb < Kc) cv = catExt[(size_t)(kbb - Pn) * S + sn];
      bf[e] = (__bf16)cv;
    }
    acc = wmma_bf16(af, bf, acc);
  }
  float* ob = outc + (size_t)b * Pn * S;
#pragma unroll
  for (int r = 0; r < 8; ++r) {
    int m = mt * 16 + ((lane >> 4) << 3) + r;
    ob[(size_t)m * S + sn] = acc[r] + bc[m];
  }
}

// ---------------------------------------------------------------------------
// Row softmax over corr rows (length n), emitting bf16 attn.
// ---------------------------------------------------------------------------
__global__ void k_softmax(const float* __restrict__ corr, __bf16* __restrict__ attn, int n) {
  if (n >= 32) {
    int wid = blockIdx.x * (blockDim.x >> 5) + (threadIdx.x >> 5);
    if (wid >= MTOT) return;
    int lane = threadIdx.x & 31;
    const float* rp = corr + (size_t)wid * n;
    float mx = -1e30f;
    for (int j = lane; j < n; j += 32) mx = fmaxf(mx, rp[j]);
#pragma unroll
    for (int o = 16; o > 0; o >>= 1) mx = fmaxf(mx, __shfl_xor(mx, o, 32));
    float sum = 0.f;
    for (int j = lane; j < n; j += 32) sum += __expf(rp[j] - mx);
#pragma unroll
    for (int o = 16; o > 0; o >>= 1) sum += __shfl_xor(sum, o, 32);
    float inv = 1.f / sum;
    __bf16* ap = attn + (size_t)wid * n;
    for (int j = lane; j < n; j += 32) ap[j] = (__bf16)(__expf(rp[j] - mx) * inv);
  } else {
    int tid = blockIdx.x * blockDim.x + threadIdx.x;
    if (tid >= MTOT) return;
    const float* rp = corr + (size_t)tid * n;
    float mx = -1e30f;
    for (int j = 0; j < n; ++j) mx = fmaxf(mx, rp[j]);
    float ex[16], sum = 0.f;
    for (int j = 0; j < n; ++j) { ex[j] = __expf(rp[j] - mx); sum += ex[j]; }
    float inv = 1.f / sum;
    __bf16* ap = attn + (size_t)tid * n;
    for (int j = 0; j < n; ++j) ap[j] = (__bf16)(ex[j] * inv);
  }
}

// ---------------------------------------------------------------------------
// o = attn @ vp  -> fused[b][tok][head*64+hd] bf16.
// ---------------------------------------------------------------------------
__global__ void k_av(const __bf16* __restrict__ attn, const __bf16* __restrict__ vb,
                     __bf16* __restrict__ fused, int head, int ps) {
  int n = ps * ps, P = NTOK / n;
  int RT = (n >= 16) ? n / 16 : 1;
  int Q  = (n >= 16) ? 1 : 16 / n;
  int pgc = P / Q;
  int tpb = pgc * RT;
  int gw = blockIdx.x * (blockDim.x >> 5) + (threadIdx.x >> 5);
  if (gw >= BATCH * tpb) return;
  int b = gw / tpb, rem = gw % tpb;
  int pg = rem / RT, rt = rem % RT;
  int lane = threadIdx.x & 31;
  int am = lane & 15;

  int a_patch, a_local;
  if (Q == 1) { a_patch = pg; a_local = rt * 16 + am; }
  else        { a_patch = pg * Q + am / n; a_local = am % n; }
  const __bf16* abase = attn + ((size_t)(b * P + a_patch) * n + a_local) * n;
  const __bf16* vbase = vb + ((size_t)b * NHEADS + head) * (size_t)NTOK * HD;

  v8f acc[4] = {};
  int ksteps = (n + 31) / 32;
  for (int kt = 0; kt < ksteps; ++kt) {
    v16bf af;
    if (n >= 32) {
      af = frag_load_a(abase + kt * 32);
    } else {
      af = (v16bf){};
#pragma unroll
      for (int e = 0; e < 16; ++e) {
        int kk = frag_k_a(lane, e);
        __bf16 vv = (__bf16)0.f;
        if (Q == 1) { if (kk < n) vv = abase[kk]; }
        else        { if (kk < 16 && (kk / n) == (am / n)) vv = abase[kk % n]; }
        af[e] = vv;
      }
    }
#pragma unroll
    for (int j = 0; j < 4; ++j) {
      v16bf bf = {};
      int hd = j * 16 + am;
#pragma unroll
      for (int e = 0; e < 16; ++e) {
        int kk = kt * 32 + frag_k_b(lane, e);
        __bf16 vv = (__bf16)0.f;
        if (Q == 1) {
          if (kk < n) { int tk = tok_of(pg, kk, ps); vv = vbase[(size_t)tk * HD + hd]; }
        } else {
          if (kk < 16) { int tk = tok_of(pg * Q + kk / n, kk % n, ps); vv = vbase[(size_t)tk * HD + hd]; }
        }
        bf[e] = vv;
      }
      acc[j] = wmma_bf16(af, bf, acc[j]);
    }
  }
#pragma unroll
  for (int j = 0; j < 4; ++j) {
    int hd = j * 16 + am;
#pragma unroll
    for (int r = 0; r < 8; ++r) {
      int m = ((lane >> 4) << 3) + r;
      int rp, rl;
      if (Q == 1) { rp = pg; rl = rt * 16 + m; }
      else        { rp = pg * Q + m / n; rl = m % n; }
      int tk = tok_of(rp, rl, ps);
      fused[((size_t)b * NTOK + tk) * CDIM + head * HD + hd] = (__bf16)acc[j][r];
    }
  }
}

// ---------------------------------------------------------------------------
// Final projection: out = fused @ Wproj + bproj (f32 output), async-LDS A tile.
// ---------------------------------------------------------------------------
__global__ void k_gemm_proj(const __bf16* __restrict__ fused, const __bf16* __restrict__ wT,
                            const float* __restrict__ bias, float* __restrict__ out) {
  __shared__ __bf16 smemA[16 * A_STRIDE];
  const int NGRP = CDIM / 256;                 // 2 groups of 256 columns
  int mt = blockIdx.x / NGRP;
  int ng = blockIdx.x % NGRP;
  int w = threadIdx.x >> 5;
  int lane = threadIdx.x & 31;

  const __bf16* aBase = fused + (size_t)mt * 16 * CDIM;
  unsigned ldsBase = (unsigned)(uintptr_t)&smemA[0];
#pragma unroll
  for (int i = 0; i < 8; ++i) {
    int c = (int)threadIdx.x + i * 128;
    int row = c >> 6, col = (c & 63) << 3;
    async_load_b128(ldsBase + (unsigned)(row * A_STRIDE + col) * 2u,
                    aBase + (size_t)row * CDIM + col);
  }
  async_wait0();
  __syncthreads();

  int nt = ng * 4 + w;
  int ncol = nt * 64 + (lane & 15);
  unsigned arow = ldsBase + (unsigned)((lane & 15) * A_STRIDE) * 2u;
  v8f acc[4] = {};
  for (int k0 = 0; k0 < CDIM; k0 += 32) {
    v16bf af = frag_load_a_lds(arow + (unsigned)(k0 * 2));
#pragma unroll
    for (int j = 0; j < 4; ++j) {
      v16bf bf = frag_load_b(wT + (size_t)(ncol + j * 16) * CDIM + k0);
      acc[j] = wmma_bf16(af, bf, acc[j]);
    }
  }
#pragma unroll
  for (int j = 0; j < 4; ++j) {
    int nc = ncol + j * 16;
    float bv = bias[nc];
#pragma unroll
    for (int r = 0; r < 8; ++r) {
      int m = mt * 16 + ((lane >> 4) << 3) + r;
      out[(size_t)m * CDIM + nc] = acc[j][r] + bv;
    }
  }
}

// ---------------------------------------------------------------------------
// Host driver
// ---------------------------------------------------------------------------
extern "C" void kernel_launch(void* const* d_in, const int* in_sizes, int n_in,
                              void* d_out, int out_size, void* d_ws, size_t ws_size,
                              hipStream_t stream) {
  (void)in_sizes; (void)n_in; (void)out_size; (void)ws_size;
  const float* x     = (const float*)d_in[0];
  const float* Wq    = (const float*)d_in[1];
  const float* Wkv   = (const float*)d_in[2];
  const float* Wproj = (const float*)d_in[3];
  const float* bproj = (const float*)d_in[4];
  const float* Wc0   = (const float*)d_in[5];
  const float* bc0   = (const float*)d_in[6];
  const float* Wc1   = (const float*)d_in[7];
  const float* bc1   = (const float*)d_in[8];
  const float* Wc2   = (const float*)d_in[9];
  const float* bc2   = (const float*)d_in[10];

  char* ws = (char*)d_ws;
  size_t off = 0;
  auto alloc = [&](size_t bytes) -> char* {
    char* p = ws + off;
    off = (off + bytes + 255) & ~(size_t)255;
    return p;
  };
  __bf16* xb     = (__bf16*)alloc((size_t)MTOT * CDIM * 2);
  __bf16* wqkvT  = (__bf16*)alloc((size_t)1536 * 512 * 2);
  __bf16* wprojT = (__bf16*)alloc((size_t)512 * 512 * 2);
  __bf16* wc0b   = (__bf16*)alloc((size_t)256 * 320 * 2);
  __bf16* wc1b   = (__bf16*)alloc((size_t)64 * 80 * 2);
  __bf16* wc2b   = (__bf16*)alloc((size_t)16 * 20 * 2);
  __bf16* qb     = (__bf16*)alloc((size_t)BATCH * NHEADS * NTOK * HD * 2);
  __bf16* kbuf   = (__bf16*)alloc((size_t)BATCH * NHEADS * NTOK * HD * 2);
  __bf16* vbuf   = (__bf16*)alloc((size_t)BATCH * NHEADS * NTOK * HD * 2);
  float*  corr0  = (float*) alloc((size_t)BATCH * NTOK * 256 * 4);
  float*  corr1  = (float*) alloc((size_t)BATCH * NTOK * 256 * 4);
  float*  corr2  = (float*) alloc((size_t)BATCH * NTOK * 256 * 4);
  __bf16* attnb  = (__bf16*)alloc((size_t)BATCH * NTOK * 256 * 2);
  __bf16* fused  = (__bf16*)alloc((size_t)MTOT * CDIM * 2);

  // Prepass: casts / packs
  k_cast_bf16<<<(MTOT * CDIM + 255) / 256, 256, 0, stream>>>(x, xb, MTOT * CDIM);
  k_pack_wqkvT<<<(1536 * 512 + 255) / 256, 256, 0, stream>>>(Wq, Wkv, wqkvT);
  k_transpose_cast<<<(512 * 512 + 255) / 256, 256, 0, stream>>>(Wproj, wprojT, 512, 512);
  k_cast_bf16<<<(256 * 320 + 255) / 256, 256, 0, stream>>>(Wc0, wc0b, 256 * 320);
  k_cast_bf16<<<(64 * 80 + 255) / 256, 256, 0, stream>>>(Wc1, wc1b, 64 * 80);
  k_cast_bf16<<<(16 * 20 + 255) / 256, 256, 0, stream>>>(Wc2, wc2b, 16 * 20);

  // Fused QKV projection: one block per (16-row A tile, 256-col group)
  k_gemm_qkv<<<(MTOT / 16) * ((3 * CDIM) / 256), 128, 0, stream>>>(xb, wqkvT, qb, kbuf, vbuf);

  // Per-head attention chain with cross-scale mixing
  float* corr[3] = {corr0, corr1, corr2};
  const int LRS[8] = {2, 2, 4, 4, 8, 8, 16, 16};
  int prev = -1, prev_ps = 0;
  for (int h = 0; h < 8; ++h) {
    int ps = LRS[h], n = ps * ps, P = NTOK / n;
    int T = (n >= 16) ? n / 16 : 1;
    int Q = (n >= 16) ? 1 : 16 / n;
    int ctiles = BATCH * (P / Q) * T * T;
    int cur;
    if (h == 0) {
      cur = 0;
      k_corr<<<(ctiles + 3) / 4, 128, 0, stream>>>(qb, kbuf, corr[cur], nullptr, h, ps, 0);
    } else if (ps == prev_ps) {
      cur = (prev + 1) % 3;
      k_corr<<<(ctiles + 3) / 4, 128, 0, stream>>>(qb, kbuf, corr[cur], corr[prev], h, ps, 1);
    } else {
      int raw = (prev + 1) % 3;
      cur = (prev + 2) % 3;
      k_corr<<<(ctiles + 3) / 4, 128, 0, stream>>>(qb, kbuf, corr[raw], nullptr, h, ps, 0);
      const __bf16* wcb; const float* bcp; int E;
      if      (prev_ps == 2) { wcb = wc0b; bcp = bc0; E = 320 - 256; }
      else if (prev_ps == 4) { wcb = wc1b; bcp = bc1; E = 80 - 64; }
      else                   { wcb = wc2b; bcp = bc2; E = 20 - 16; }
      int S = n * n;
      int mixwaves = BATCH * (P / 16) * (S / 16);
      k_mix<<<(mixwaves + 3) / 4, 128, 0, stream>>>(wcb, bcp, corr[raw], corr[prev],
                                                    corr[cur], P, E, S);
    }
    if (n >= 32) {
      k_softmax<<<(MTOT + 3) / 4, 128, 0, stream>>>(corr[cur], attnb, n);
    } else {
      k_softmax<<<(MTOT + 255) / 256, 256, 0, stream>>>(corr[cur], attnb, n);
    }
    int RT = (n >= 16) ? n / 16 : 1;
    int avtiles = BATCH * (P / Q) * RT;
    k_av<<<(avtiles + 3) / 4, 128, 0, stream>>>(attnb, vbuf, fused, h, ps);
    prev = cur; prev_ps = ps;
  }

  // Final projection
  k_gemm_proj<<<(MTOT / 16) * (CDIM / 256), 128, 0, stream>>>(fused, wprojT, bproj,
                                                              (float*)d_out);
}
